// CrossAttentionBlock_63273458204772
// MI455X (gfx1250) — matrix operand order repaired and checked by reading, not compiled, not run on hardware
//
#include <hip/hip_runtime.h>
#include <hip/hip_bf16.h>

#define DEVINL __device__ __forceinline__

typedef __attribute__((ext_vector_type(16))) __bf16       v16bf;
typedef __attribute__((ext_vector_type(8)))  float        v8f;
typedef __attribute__((ext_vector_type(8)))  unsigned int v8u;

constexpr int cH  = 2048;
constexpr int cNH = 16;
constexpr int cHD = 128;
constexpr int cB  = 4;
constexpr int cS  = 2048;
constexpr int cT  = 256;

// ---------- scalar helpers ----------
DEVINL unsigned short f2bf(float f) {
  unsigned int u = __builtin_bit_cast(unsigned int, f);
  u += 0x7FFFu + ((u >> 16) & 1u);          // round-to-nearest-even
  return (unsigned short)(u >> 16);
}
DEVINL unsigned int pack2(float lo, float hi) {
  return (unsigned int)f2bf(lo) | ((unsigned int)f2bf(hi) << 16);
}
// branch-free nan_to_num(nan->0, inf->+/-1e4); finite values untouched
DEVINL float nclean(float x) {
  unsigned int u = __builtin_bit_cast(unsigned int, x);
  unsigned int a = u & 0x7fffffffu;
  float lim = (u >> 31) ? -10000.0f : 10000.0f;
  float r = (a > 0x7f800000u) ? 0.0f : x;   // NaN -> 0
  r = (a == 0x7f800000u) ? lim : r;         // Inf -> +/-1e4
  return r;
}
DEVINL v8f vzero() {
  v8f v;
#pragma unroll
  for (int i = 0; i < 8; i++) v[i] = 0.0f;
  return v;
}

DEVINL v8f wmma_bf16(v8u a, v8u b, v8f c) {
  return __builtin_amdgcn_wmma_f32_16x16x32_bf16(
      false, __builtin_bit_cast(v16bf, a),
      false, __builtin_bit_cast(v16bf, b),
      (short)0, c, false, false);
}

// ---------- fragment loaders (ISA 7.12.2 layouts, wave32) ----------
// A 16x32 bf16 row-major source, lane: m = lane&15, g = lane>>4.
// v0..v3 hold K = k0+g*8+{0..7}; v4..v7 hold K = k0+16+g*8+{0..7} (pairs per VGPR).
DEVINL v8u load_a_bf16(const unsigned short* X, int ld, int row0, int k0, int m, int g) {
  const unsigned int* p = (const unsigned int*)(X + (size_t)(row0 + m) * ld);
  int base = (k0 + g * 8) >> 1;
  v8u r;
  r[0] = p[base + 0]; r[1] = p[base + 1]; r[2] = p[base + 2]; r[3] = p[base + 3];
  r[4] = p[base + 8]; r[5] = p[base + 9]; r[6] = p[base + 10]; r[7] = p[base + 11];
  return r;
}
// B 32x16: B[k][n] = Wrow[n][k] where Wrow is (N x Kld) row-major (weight, K rows, or Vt rows).
// lane: n = lane&15, g = lane>>4; v0..v3 hold K = k0+g*8+{0..7}; v4..v7 hold K = k0+16+g*8+{0..7}.
DEVINL v8u load_b_rows(const unsigned short* W, int ld, int col0, int k0, int n, int g) {
  const unsigned int* p = (const unsigned int*)(W + (size_t)(col0 + n) * ld);
  int base = (k0 + g * 8) >> 1;
  v8u r;
#pragma unroll
  for (int i = 0; i < 8; i++) r[i] = p[base + i + (i >> 2) * 4];
  return r;
}

// ---------- conversion kernels (vectorized, branch-free) ----------
__global__ __launch_bounds__(256) void cvt_w_kernel(
    const float4* __restrict__ src, uint2* __restrict__ dst, int n4) {
  int i = blockIdx.x * 256 + threadIdx.x;
  if (i < n4) {
    float4 f = src[i];
    dst[i] = make_uint2(pack2(f.x, f.y), pack2(f.z, f.w));
  }
}
__global__ __launch_bounds__(256) void cvt_act_kernel(
    const float4* __restrict__ src, uint2* __restrict__ dst, int n4) {
  int i = blockIdx.x * 256 + threadIdx.x;
  if (i < n4) {
    float4 f = src[i];
    dst[i] = make_uint2(pack2(nclean(f.x), nclean(f.y)),
                        pack2(nclean(f.z), nclean(f.w)));
  }
}

// ---------- GEMM: Y(bf16) = X(bf16) @ W^T + b ----------
// block = 256 thr (8 waves), tile 128(M) x 128(N); wave = 32x64 (2x4 subtiles)
__global__ __launch_bounds__(256) void gemm_qkv_kernel(
    const unsigned short* __restrict__ X, const unsigned short* __restrict__ W,
    const float* __restrict__ bias, unsigned short* __restrict__ Y,
    int M, int K, int N) {
  const int lane = threadIdx.x & 31;
  const int wave = threadIdx.x >> 5;
  const int m_ = lane & 15, g = lane >> 4;
  const int wm = wave >> 1, wn = wave & 1;
  const int row0 = blockIdx.x * 128 + wm * 32;
  const int col0 = blockIdx.y * 128 + wn * 64;

  v8f acc[2][4];
#pragma unroll
  for (int a = 0; a < 2; a++)
#pragma unroll
    for (int b = 0; b < 4; b++) acc[a][b] = vzero();

  for (int k0 = 0; k0 < K; k0 += 32) {
    if (k0 + 64 < K)  // gfx1250 global_prefetch_b8 path
      __builtin_prefetch(X + (size_t)(row0 + m_) * K + k0 + 64, 0, 3);
    v8u a0 = load_a_bf16(X, K, row0, k0, m_, g);
    v8u a1 = load_a_bf16(X, K, row0 + 16, k0, m_, g);
#pragma unroll
    for (int nt = 0; nt < 4; nt++) {
      v8u b = load_b_rows(W, K, col0 + nt * 16, k0, m_, g);
      acc[0][nt] = wmma_bf16(a0, b, acc[0][nt]);
      acc[1][nt] = wmma_bf16(a1, b, acc[1][nt]);
    }
  }
#pragma unroll
  for (int mt = 0; mt < 2; mt++)
#pragma unroll
    for (int nt = 0; nt < 4; nt++) {
      int col = col0 + nt * 16 + m_;
      float bv = bias[col];
#pragma unroll
      for (int i = 0; i < 8; i++) {
        int row = row0 + mt * 16 + i + 8 * g;
        Y[(size_t)row * N + col] = f2bf(acc[mt][nt][i] + bv);
      }
    }
}

// ---------- per-head transpose of V -> Vt[b][h][d][t] ----------
__global__ __launch_bounds__(256) void transpose_v_kernel(
    const unsigned short* __restrict__ V, unsigned short* __restrict__ Vt) {
  int idx = blockIdx.x * 256 + threadIdx.x;
  if (idx >= cB * cT * cH) return;
  int col = idx % cH, row = idx / cH;
  int h = col / cHD, d = col % cHD;
  int t = row % cT, b = row / cT;
  Vt[(((size_t)b * cNH + h) * cHD + d) * cT + t] = V[idx];
}

// ---------- attention (scores -> mask/softmax -> P@V) ----------
// block = 128 thr (4 waves); wave handles 16 S-rows for one (b,h); T=256 in regs
__global__ __launch_bounds__(128) void attn_kernel(
    const unsigned short* __restrict__ Q, const unsigned short* __restrict__ Kb,
    const unsigned short* __restrict__ Vt, const int* __restrict__ mask,
    unsigned short* __restrict__ Ctx) {
  __shared__ unsigned short Plds[4][16 * cT];  // 32 KB
  const int lane = threadIdx.x & 31;
  const int wave = threadIdx.x >> 5;
  const int nl = lane & 15, g = lane >> 4;
  const int b = blockIdx.z, h = blockIdx.y;
  const int s0 = blockIdx.x * 64 + wave * 16;
  const int hcol = h * cHD;

  // scores: 16 C-tiles covering T=256
  v8f sc[16];
#pragma unroll
  for (int j = 0; j < 16; j++) sc[j] = vzero();
  for (int kk = 0; kk < cHD; kk += 32) {
    v8u aq = load_a_bf16(Q, cH, b * cS + s0, hcol + kk, nl, g);
#pragma unroll
    for (int j = 0; j < 16; j++) {
      v8u bk = load_b_rows(Kb, cH, b * cT + j * 16, hcol + kk, nl, g);
      sc[j] = wmma_bf16(aq, bk, sc[j]);
    }
  }
  // mask bits for this lane's column (t = j*16 + nl)
  unsigned attb = 0;
#pragma unroll
  for (int j = 0; j < 16; j++)
    if (mask[b * cT + j * 16 + nl] > 0) attb |= (1u << j);

  const float scale = 0.08838834764831845f;  // 1/sqrt(128)
  float rmax[8], rsum[8];
#pragma unroll
  for (int i = 0; i < 8; i++) rmax[i] = -1e30f;
#pragma unroll
  for (int j = 0; j < 16; j++)
#pragma unroll
    for (int i = 0; i < 8; i++) {
      float s = sc[j][i] * scale;
      s = fminf(fmaxf(s, -50.0f), 50.0f);
      s = ((attb >> j) & 1u) ? s : -50.0f;  // -10000 then clip -> -50
      sc[j][i] = s;
      rmax[i] = fmaxf(rmax[i], s);
    }
  // cross-lane reductions stay inside each 16-lane half (C layout rows i+8g)
#pragma unroll
  for (int off = 1; off < 16; off <<= 1)
#pragma unroll
    for (int i = 0; i < 8; i++) rmax[i] = fmaxf(rmax[i], __shfl_xor(rmax[i], off, 32));
#pragma unroll
  for (int i = 0; i < 8; i++) rsum[i] = 0.0f;
#pragma unroll
  for (int j = 0; j < 16; j++)
#pragma unroll
    for (int i = 0; i < 8; i++) {
      float e = __expf(sc[j][i] - rmax[i]);
      sc[j][i] = e;
      rsum[i] += e;
    }
#pragma unroll
  for (int off = 1; off < 16; off <<= 1)
#pragma unroll
    for (int i = 0; i < 8; i++) rsum[i] += __shfl_xor(rsum[i], off, 32);
  float rinv[8];
#pragma unroll
  for (int i = 0; i < 8; i++) rinv[i] = 1.0f / (rsum[i] * (1.0f + 1e-8f));

  // write P (bf16) to this wave's LDS region, row-major 16 x 256
  unsigned short* Pw = &Plds[wave][0];
#pragma unroll
  for (int j = 0; j < 16; j++)
#pragma unroll
    for (int i = 0; i < 8; i++)
      Pw[(i + 8 * g) * cT + j * 16 + nl] = f2bf(sc[j][i] * rinv[i]);
  __syncthreads();

  // ctx = P @ V  (K=T=256, N=HD=128 -> 8 d-tiles)
  v8f ct[8];
#pragma unroll
  for (int d = 0; d < 8; d++) ct[d] = vzero();
  for (int kk = 0; kk < cT; kk += 32) {
    v8u ap = load_a_bf16(Pw, cT, 0, kk, nl, g);
#pragma unroll
    for (int dt = 0; dt < 8; dt++) {
      v8u bv = load_b_rows(Vt, cT, (b * cNH + h) * cHD + dt * 16, kk, nl, g);
      ct[dt] = wmma_bf16(ap, bv, ct[dt]);
    }
  }
#pragma unroll
  for (int dt = 0; dt < 8; dt++)
#pragma unroll
    for (int i = 0; i < 8; i++)
      Ctx[(size_t)(b * cS + s0 + i + 8 * g) * cH + hcol + dt * 16 + nl] =
          f2bf(ct[dt][i]);
}

// ---------- delta(f32) = rs * nan_clean(ctx(bf16) @ Wo^T + bo) ----------
__global__ __launch_bounds__(256) void gemm_out_kernel(
    const unsigned short* __restrict__ X, const unsigned short* __restrict__ W,
    const float* __restrict__ bias, const float* __restrict__ rs_ptr,
    float* __restrict__ Y, int M, int K, int N) {
  const int lane = threadIdx.x & 31;
  const int wave = threadIdx.x >> 5;
  const int m_ = lane & 15, g = lane >> 4;
  const int wm = wave >> 1, wn = wave & 1;
  const int row0 = blockIdx.x * 128 + wm * 32;
  const int col0 = blockIdx.y * 128 + wn * 64;
  const float rs = fminf(fmaxf(rs_ptr[0], 0.0f), 0.3f);

  v8f acc[2][4];
#pragma unroll
  for (int a = 0; a < 2; a++)
#pragma unroll
    for (int b = 0; b < 4; b++) acc[a][b] = vzero();

  for (int k0 = 0; k0 < K; k0 += 32) {
    if (k0 + 64 < K)
      __builtin_prefetch(X + (size_t)(row0 + m_) * K + k0 + 64, 0, 3);
    v8u a0 = load_a_bf16(X, K, row0, k0, m_, g);
    v8u a1 = load_a_bf16(X, K, row0 + 16, k0, m_, g);
#pragma unroll
    for (int nt = 0; nt < 4; nt++) {
      v8u b = load_b_rows(W, K, col0 + nt * 16, k0, m_, g);
      acc[0][nt] = wmma_bf16(a0, b, acc[0][nt]);
      acc[1][nt] = wmma_bf16(a1, b, acc[1][nt]);
    }
  }
#pragma unroll
  for (int mt = 0; mt < 2; mt++)
#pragma unroll
    for (int nt = 0; nt < 4; nt++) {
      int col = col0 + nt * 16 + m_;
      float bv = bias[col];
#pragma unroll
      for (int i = 0; i < 8; i++) {
        int row = row0 + mt * 16 + i + 8 * g;
        Y[(size_t)row * N + col] = rs * nclean(acc[mt][nt][i] + bv);
      }
    }
}

// ---------- row LayerNorm (biased var) ----------
__global__ __launch_bounds__(256) void layernorm_kernel(
    const float* __restrict__ delta, const float* __restrict__ gamma,
    const float* __restrict__ beta, float* __restrict__ out) {
  const int row = blockIdx.x;
  const float* x = delta + (size_t)row * cH;
  float lsum = 0.0f, lsq = 0.0f;
  for (int c = threadIdx.x; c < cH; c += 256) {
    float v = x[c];
    lsum += v;
    lsq += v * v;
  }
  __shared__ float s1[256], s2[256];
  s1[threadIdx.x] = lsum;
  s2[threadIdx.x] = lsq;
  __syncthreads();
  for (int st = 128; st > 0; st >>= 1) {
    if ((int)threadIdx.x < st) {
      s1[threadIdx.x] += s1[threadIdx.x + st];
      s2[threadIdx.x] += s2[threadIdx.x + st];
    }
    __syncthreads();
  }
  const float mu = s1[0] * (1.0f / cH);
  const float var = s2[0] * (1.0f / cH) - mu * mu;
  const float inv = rsqrtf(var + 1e-5f);
  for (int c = threadIdx.x; c < cH; c += 256) {
    float v = (x[c] - mu) * inv;
    out[(size_t)row * cH + c] = v * gamma[c] + beta[c];
  }
}

// ---------- host launch ----------
extern "C" void kernel_launch(void* const* d_in, const int* in_sizes, int n_in,
                              void* d_out, int out_size, void* d_ws, size_t ws_size,
                              hipStream_t stream) {
  (void)in_sizes; (void)n_in; (void)out_size; (void)ws_size;
  const float* hs = (const float*)d_in[0];
  const float* at = (const float*)d_in[1];
  const int*   am = (const int*)d_in[2];
  const float* Wq = (const float*)d_in[3];
  const float* bq = (const float*)d_in[4];
  const float* Wk = (const float*)d_in[5];
  const float* bk = (const float*)d_in[6];
  const float* Wv = (const float*)d_in[7];
  const float* bv = (const float*)d_in[8];
  const float* Wo = (const float*)d_in[9];
  const float* bo = (const float*)d_in[10];
  const float* gam = (const float*)d_in[11];
  const float* bet = (const float*)d_in[12];
  const float* rs  = (const float*)d_in[13];
  float* out = (float*)d_out;

  char* ws = (char*)d_ws;
  size_t off = 0;
  auto carve = [&](size_t bytes) {
    char* p = ws + off;
    off += (bytes + 255) & ~(size_t)255;
    return p;
  };
  const size_t wsz  = (size_t)cH * cH;       // weight elems
  const size_t qsz  = (size_t)cB * cS * cH;  // hs / Q / ctx elems
  const size_t kvsz = (size_t)cB * cT * cH;  // at / K / V elems
  unsigned short* Wqb = (unsigned short*)carve(wsz * 2);
  unsigned short* Wkb = (unsigned short*)carve(wsz * 2);
  unsigned short* Wvb = (unsigned short*)carve(wsz * 2);
  unsigned short* Wob = (unsigned short*)carve(wsz * 2);
  unsigned short* Hsb = (unsigned short*)carve(qsz * 2);
  unsigned short* Atb = (unsigned short*)carve(kvsz * 2);
  unsigned short* Qb  = (unsigned short*)carve(qsz * 2);
  unsigned short* Kbf = (unsigned short*)carve(kvsz * 2);
  unsigned short* Vbf = (unsigned short*)carve(kvsz * 2);
  unsigned short* Vtb = (unsigned short*)carve(kvsz * 2);
  unsigned short* Ctx = (unsigned short*)carve(qsz * 2);
  float*          dlt = (float*)carve(qsz * 4);

  const int M_q = cB * cS;   // 8192
  const int M_kv = cB * cT;  // 1024

  // weights / activations -> bf16 (activations get nan_clean)
  cvt_w_kernel<<<(int)(wsz / 4 / 256), 256, 0, stream>>>((const float4*)Wq, (uint2*)Wqb, (int)(wsz / 4));
  cvt_w_kernel<<<(int)(wsz / 4 / 256), 256, 0, stream>>>((const float4*)Wk, (uint2*)Wkb, (int)(wsz / 4));
  cvt_w_kernel<<<(int)(wsz / 4 / 256), 256, 0, stream>>>((const float4*)Wv, (uint2*)Wvb, (int)(wsz / 4));
  cvt_w_kernel<<<(int)(wsz / 4 / 256), 256, 0, stream>>>((const float4*)Wo, (uint2*)Wob, (int)(wsz / 4));
  cvt_act_kernel<<<(int)(qsz / 4 / 256), 256, 0, stream>>>((const float4*)hs, (uint2*)Hsb, (int)(qsz / 4));
  cvt_act_kernel<<<(int)(kvsz / 4 / 256), 256, 0, stream>>>((const float4*)at, (uint2*)Atb, (int)(kvsz / 4));

  // projections
  dim3 gq(M_q / 128, cH / 128);
  gemm_qkv_kernel<<<gq, 256, 0, stream>>>(Hsb, Wqb, bq, Qb, M_q, cH, cH);
  dim3 gkv(M_kv / 128, cH / 128);
  gemm_qkv_kernel<<<gkv, 256, 0, stream>>>(Atb, Wkb, bk, Kbf, M_kv, cH, cH);
  gemm_qkv_kernel<<<gkv, 256, 0, stream>>>(Atb, Wvb, bv, Vbf, M_kv, cH, cH);

  // V transpose per head
  transpose_v_kernel<<<(int)(kvsz / 256), 256, 0, stream>>>(Vbf, Vtb);

  // attention
  attn_kernel<<<dim3(cS / 64, cNH, cB), 128, 0, stream>>>(Qb, Kbf, Vtb, am, Ctx);

  // output projection + residual scale
  gemm_out_kernel<<<gq, 256, 0, stream>>>(Ctx, Wob, bo, rs, dlt, M_q, cH, cH);

  // layernorm
  layernorm_kernel<<<M_q, 256, 0, stream>>>(dlt, gam, bet, out);
}
